// MultiHeadAttention_16741782520380
// MI455X (gfx1250) — matrix operand order repaired and checked by reading, not compiled, not run on hardware
//
#include <hip/hip_runtime.h>

// ---------------------------------------------------------------------------
// MHA forward for MI455X (gfx1250, wave32, WMMA).
// All GEMMs: C[m,n] = sum_k A[m,k] * Bk[k,n]  (Bk k-major -> b128-only staging).
// 128x64 block tile, 64x32 wave tile (8 WMMA / 8 ds_load_b128 + 32 16-bit LDS loads per K-step),
// double-buffered LDS, 1 barrier per K-step.
// bf16-A GEMMs: GLOBAL_LOAD_ASYNC_TO_LDS_B128 staging (ASYNCcnt), no VGPR
// round-trip. fp32-A GEMM (attn@V): register pipeline with on-the-fly cast.
// ---------------------------------------------------------------------------

typedef __attribute__((ext_vector_type(16))) __bf16 v16bf;
typedef __attribute__((ext_vector_type(8)))  __bf16 v8bf;
typedef __attribute__((ext_vector_type(8)))  float  v8f;

constexpr int D_MODEL = 1024;
constexpr int NHEADS  = 16;
constexpr int DK      = 64;
constexpr int BATCH   = 2;
constexpr int SEQ     = 2048;
constexpr int MTOT    = BATCH * SEQ;

constexpr int BM = 128;      // block tile M
constexpr int BN = 64;       // block tile N
constexpr int BK = 32;       // K-step
constexpr int APITCH = 40;   // A tile pitch (80B rows, 16B aligned)
constexpr int BPITCH = 72;   // B tile pitch (144B rows, 16B aligned)
constexpr int AELEMS = BM * APITCH;
constexpr int BELEMS = BK * BPITCH;

// ---------------------------------------------------------------------------
__global__ void cast_bf16_kernel(const float* __restrict__ in,
                                 __bf16* __restrict__ out, int n) {
  int i = blockIdx.x * blockDim.x + threadIdx.x;
  if (i < n) out[i] = (__bf16)in[i];
}

// Tiled transpose+cast: out[k*rows + n] = (bf16) in[n*cols + k]
__global__ __launch_bounds__(256) void transpose_cast_kernel(
    const float* __restrict__ in, __bf16* __restrict__ out, int rows, int cols) {
  __shared__ float tile[32][33];
  const int bx = blockIdx.x * 32;
  const int by = blockIdx.y * 32;
  const int tx = threadIdx.x & 31;
  const int ty = (threadIdx.x >> 5) * 4;
#pragma unroll
  for (int j = 0; j < 4; ++j)
    tile[ty + j][tx] = in[(size_t)(by + ty + j) * cols + bx + tx];
  __syncthreads();
#pragma unroll
  for (int j = 0; j < 4; ++j)
    out[(size_t)(bx + ty + j) * rows + by + tx] = (__bf16)tile[tx][ty + j];
}

// ---------------------------------------------------------------------------
// Fragment loaders (ISA 7.12.2 layouts)
// ---------------------------------------------------------------------------
static __device__ __forceinline__ v16bf load_a_frag(const __bf16* As, int rowBase) {
  const int lane = threadIdx.x & 31;
  const int m  = rowBase + (lane & 15);
  const int kb = (lane >> 4) << 3;
  const __bf16* p = As + m * APITCH + kb;
  union { v16bf v; v8bf h[2]; } u;
  u.h[0] = *(const v8bf*)(p);
  u.h[1] = *(const v8bf*)(p + 16);
  return u.v;
}

static __device__ __forceinline__ v16bf load_b_frag(const __bf16* Bs, int colBase) {
  const int lane = threadIdx.x & 31;
  const int kb = (lane >> 4) << 3;
  const __bf16* p = Bs + kb * BPITCH + colBase + (lane & 15);
  v16bf v;
#pragma unroll
  for (int s = 0; s < 8; ++s) {
    v[s]     = p[s * BPITCH];
    v[s + 8] = p[(16 + s) * BPITCH];
  }
  return v;
}

// 8 WMMAs for one 64x32 wave tile against one staged K-step
static __device__ __forceinline__ void compute_step(
    v8f (&c)[4][2], const __bf16* Ab, const __bf16* Bb, int waveM, int waveN) {
  const v16bf b0 = load_b_frag(Bb, waveN + 0);
  const v16bf b1 = load_b_frag(Bb, waveN + 16);
#pragma unroll
  for (int fr = 0; fr < 4; ++fr) {
    const v16bf a = load_a_frag(Ab, waveM + fr * 16);
    c[fr][0] = __builtin_amdgcn_wmma_f32_16x16x32_bf16(false, a, false, b0, (short)0, c[fr][0], false, false);
    c[fr][1] = __builtin_amdgcn_wmma_f32_16x16x32_bf16(false, a, false, b1, (short)0, c[fr][1], false, false);
  }
}

// ---------------------------------------------------------------------------
// Async staging (bf16 A): GLOBAL_LOAD_ASYNC_TO_LDS_B128, tracked by ASYNCcnt.
// VDST = per-lane LDS byte address (low 32 bits of generic pointer),
// VADDR = 64-bit global address (ISA 15.18.3 / CDNA5 ISA).
// ---------------------------------------------------------------------------
static __device__ __forceinline__ void async_copy_b128(void* lds, const void* gptr) {
  const unsigned lds32 = (unsigned)(unsigned long long)lds;
  asm volatile("global_load_async_to_lds_b128 %0, %1, off"
               :: "v"(lds32), "v"(gptr)
               : "memory");
}
static __device__ __forceinline__ void wait_asynccnt0() {
  asm volatile("s_wait_asynccnt 0x0" ::: "memory");
}

struct AsyncStage {
  const __bf16* aP[4];
  const __bf16* bP[2];
  size_t bStep;
  int aOff[4], bOff[2];

  __device__ __forceinline__ void init(const __bf16* Abf, const __bf16* Bk,
                                       int K, int ldb, int blockM, int blockN) {
    const int tid = threadIdx.x;
#pragma unroll
    for (int i = 0; i < 4; ++i) {
      const int idx = tid + i * 128;
      const int row = idx >> 2, seg = idx & 3;
      aP[i] = Abf + (size_t)(blockM + row) * K + seg * 8;
      aOff[i] = row * APITCH + seg * 8;
    }
#pragma unroll
    for (int i = 0; i < 2; ++i) {
      const int idx = tid + i * 128;
      const int krow = idx >> 3, seg = idx & 7;
      bP[i] = Bk + (size_t)krow * ldb + blockN + seg * 8;
      bOff[i] = krow * BPITCH + seg * 8;
    }
    bStep = (size_t)BK * ldb;
  }

  __device__ __forceinline__ void issue(__bf16* As, __bf16* Bs) {
#pragma unroll
    for (int i = 0; i < 4; ++i) {
      async_copy_b128(&As[aOff[i]], aP[i]);
      __builtin_prefetch(aP[i] + 2 * BK, 0, 1);   // speculative, safe past end
      aP[i] += BK;
    }
#pragma unroll
    for (int i = 0; i < 2; ++i) {
      async_copy_b128(&Bs[bOff[i]], bP[i]);
      __builtin_prefetch(bP[i] + 2 * bStep, 0, 1);
      bP[i] += bStep;
    }
  }
};

// Register staging (fp32 A -> on-the-fly bf16 cast)
struct RegStage {
  const float* aPf[4];
  const __bf16* bP[2];
  size_t bStep;
  int aOff[4], bOff[2];
  v8bf ra[4], rb[2];

  __device__ __forceinline__ void init(const float* Af32, const __bf16* Bk,
                                       int K, int ldb, int blockM, int blockN) {
    const int tid = threadIdx.x;
#pragma unroll
    for (int i = 0; i < 4; ++i) {
      const int idx = tid + i * 128;
      const int row = idx >> 2, seg = idx & 3;
      aPf[i] = Af32 + (size_t)(blockM + row) * K + seg * 8;
      aOff[i] = row * APITCH + seg * 8;
    }
#pragma unroll
    for (int i = 0; i < 2; ++i) {
      const int idx = tid + i * 128;
      const int krow = idx >> 3, seg = idx & 7;
      bP[i] = Bk + (size_t)krow * ldb + blockN + seg * 8;
      bOff[i] = krow * BPITCH + seg * 8;
    }
    bStep = (size_t)BK * ldb;
  }

  __device__ __forceinline__ void load() {
#pragma unroll
    for (int i = 0; i < 4; ++i) {
      float4 f0 = *(const float4*)(aPf[i]);
      float4 f1 = *(const float4*)(aPf[i] + 4);
      v8bf t;
      t[0] = (__bf16)f0.x; t[1] = (__bf16)f0.y; t[2] = (__bf16)f0.z; t[3] = (__bf16)f0.w;
      t[4] = (__bf16)f1.x; t[5] = (__bf16)f1.y; t[6] = (__bf16)f1.z; t[7] = (__bf16)f1.w;
      ra[i] = t;
      __builtin_prefetch(aPf[i] + 2 * BK, 0, 1);
      aPf[i] += BK;
    }
#pragma unroll
    for (int i = 0; i < 2; ++i) {
      rb[i] = *(const v8bf*)(bP[i]);
      __builtin_prefetch(bP[i] + 2 * bStep, 0, 1);
      bP[i] += bStep;
    }
  }

  __device__ __forceinline__ void store(__bf16* As, __bf16* Bs) const {
#pragma unroll
    for (int i = 0; i < 4; ++i) *(v8bf*)&As[aOff[i]] = ra[i];
#pragma unroll
    for (int i = 0; i < 2; ++i) *(v8bf*)&Bs[bOff[i]] = rb[i];
  }
};

// ---------------------------------------------------------------------------
// Generic WMMA GEMM: A row-major MxK, Bk k-major KxN (row stride ldb = N).
// MODE 0: proj -> bf16 (B,H,S,dk), +bias             (Q and V)
// MODE 1: proj -> bf16 (B,H,dk,S), +bias             (K, pre-transposed)
// MODE 2: proj -> fp32 row-major, +bias              (final output)
// MODE 4: scores -> fp32 attn[z], *scale, mask       (batched over B*H)
// MODE 5: attn @ V -> bf16 (B,S,D) head-merge, A fp32 (batched over B*H)
// ---------------------------------------------------------------------------
template <int MODE, bool AF32>
__global__ __launch_bounds__(128) void gemm_wmma_nt(
    const void* __restrict__ Abase, const __bf16* __restrict__ Bkbase,
    const float* __restrict__ bias, const int* __restrict__ mask,
    void* __restrict__ outbase, int M, int N, int K, float scale) {
  __shared__ alignas(16) __bf16 AsS[2][AELEMS];
  __shared__ alignas(16) __bf16 BsS[2][BELEMS];

  const int tid    = threadIdx.x;
  const int lane   = tid & 31;
  const int wave   = tid >> 5;
  const int waveM  = (wave >> 1) * 64;   // 2x2 wave grid, 64x32 wave tile
  const int waveN  = (wave & 1) * 32;
  const int blockM = blockIdx.y * BM;
  const int blockN = blockIdx.x * BN;
  const int z      = blockIdx.z;

  const __bf16* Abf  = (const __bf16*)Abase;
  const float*  Af32 = (const float*)Abase;
  const __bf16* Bk   = Bkbase;
  if constexpr (MODE == 4) { Abf += (size_t)z * SEQ * DK;  Bk += (size_t)z * DK * SEQ; }
  if constexpr (MODE == 5) { Af32 += (size_t)z * SEQ * SEQ; Bk += (size_t)z * SEQ * DK; }

  v8f c[4][2] = {};
  int cur = 0;

  if constexpr (!AF32) {
    // -------- async-to-LDS pipeline --------
    AsyncStage st;
    st.init(Abf, Bk, K, N, blockM, blockN);
    st.issue(AsS[0], BsS[0]);
    wait_asynccnt0();
    __syncthreads();
    for (int k0 = 0; k0 < K; k0 += BK) {
      const bool hasNext = (k0 + BK < K);
      if (hasNext) st.issue(AsS[cur ^ 1], BsS[cur ^ 1]);   // DMA over WMMAs
      compute_step(c, AsS[cur], BsS[cur], waveM, waveN);
      if (hasNext) {
        wait_asynccnt0();
        __syncthreads();
        cur ^= 1;
      }
    }
  } else {
    // -------- register pipeline (fp32 A, cast at stage) --------
    RegStage st;
    st.init(Af32, Bk, K, N, blockM, blockN);
    st.load();
    st.store(AsS[0], BsS[0]);
    __syncthreads();
    for (int k0 = 0; k0 < K; k0 += BK) {
      const bool hasNext = (k0 + BK < K);
      if (hasNext) st.load();
      compute_step(c, AsS[cur], BsS[cur], waveM, waveN);
      if (hasNext) {
        st.store(AsS[cur ^ 1], BsS[cur ^ 1]);
        __syncthreads();
        cur ^= 1;
      }
    }
  }

  // epilogue: C layout = lane -> N(0..15), VGPR r -> M (r / r+8)
  const int cn   = lane & 15;
  const int rsel = (lane >> 4) << 3;
#pragma unroll
  for (int fr = 0; fr < 4; ++fr)
#pragma unroll
    for (int fc = 0; fc < 2; ++fc)
#pragma unroll
      for (int r = 0; r < 8; ++r) {
        const int m = blockM + waveM + fr * 16 + rsel + r;
        const int n = blockN + waveN + fc * 16 + cn;
        const float acc = c[fr][fc][r];
        if constexpr (MODE == 0 || MODE == 1) {
          const float val = acc + bias[n];
          const int b = m >> 11, s = m & (SEQ - 1);
          const int h = n >> 6,  d = n & (DK - 1);
          const size_t addr = (MODE == 0)
              ? ((((size_t)b * NHEADS + h) * SEQ + s) * DK + d)
              : ((((size_t)b * NHEADS + h) * DK + d) * SEQ + s);
          ((__bf16*)outbase)[addr] = (__bf16)val;
        } else if constexpr (MODE == 2) {
          ((float*)outbase)[(size_t)m * N + n] = acc + bias[n];
        } else if constexpr (MODE == 4) {
          const int b = z >> 4;
          const int mk = mask[((size_t)b * SEQ + m) * SEQ + n];
          const float val = (mk == 0) ? -1e9f : acc * scale;
          ((float*)outbase)[(size_t)z * SEQ * SEQ + (size_t)m * SEQ + n] = val;
        } else if constexpr (MODE == 5) {
          const int b = z >> 4, h = z & 15;
          ((__bf16*)outbase)[((size_t)b * SEQ + m) * D_MODEL + h * DK + n] = (__bf16)acc;
        }
      }
}

// ---------------------------------------------------------------------------
// Row softmax in place: one 256-thread block per 2048-wide row.
// ---------------------------------------------------------------------------
__global__ __launch_bounds__(256) void softmax_rows(float* __restrict__ attn) {
  __shared__ float red[256];
  const int tid = threadIdx.x;
  float* p = attn + (size_t)blockIdx.x * SEQ + tid * 8;
  float4 u0 = *(const float4*)(p);
  float4 u1 = *(const float4*)(p + 4);
  float v[8] = {u0.x, u0.y, u0.z, u0.w, u1.x, u1.y, u1.z, u1.w};

  float m = v[0];
#pragma unroll
  for (int j = 1; j < 8; ++j) m = fmaxf(m, v[j]);
  red[tid] = m;
  __syncthreads();
  for (int s = 128; s > 0; s >>= 1) {
    if (tid < s) red[tid] = fmaxf(red[tid], red[tid + s]);
    __syncthreads();
  }
  m = red[0];
  __syncthreads();

  float sum = 0.f;
#pragma unroll
  for (int j = 0; j < 8; ++j) { v[j] = __expf(v[j] - m); sum += v[j]; }
  red[tid] = sum;
  __syncthreads();
  for (int s = 128; s > 0; s >>= 1) {
    if (tid < s) red[tid] += red[tid + s];
    __syncthreads();
  }
  const float inv = 1.0f / red[0];
#pragma unroll
  for (int j = 0; j < 8; ++j) v[j] *= inv;
  *(float4*)(p)     = make_float4(v[0], v[1], v[2], v[3]);
  *(float4*)(p + 4) = make_float4(v[4], v[5], v[6], v[7]);
}

// ---------------------------------------------------------------------------
extern "C" void kernel_launch(void* const* d_in, const int* in_sizes, int n_in,
                              void* d_out, int out_size, void* d_ws, size_t ws_size,
                              hipStream_t stream) {
  const float* q    = (const float*)d_in[0];
  const float* k    = (const float*)d_in[1];
  const float* v    = (const float*)d_in[2];
  const int*   mask = (const int*)d_in[3];
  const float* Wq = (const float*)d_in[4];  const float* bq = (const float*)d_in[5];
  const float* Wk = (const float*)d_in[6];  const float* bk = (const float*)d_in[7];
  const float* Wv = (const float*)d_in[8];  const float* bv = (const float*)d_in[9];
  const float* Wo = (const float*)d_in[10]; const float* bo = (const float*)d_in[11];

  constexpr size_t XE = (size_t)MTOT * D_MODEL;
  constexpr size_t WE = (size_t)D_MODEL * D_MODEL;

  __bf16* qbf = (__bf16*)d_ws;
  __bf16* kbf = qbf + XE;
  __bf16* vbf = kbf + XE;
  __bf16* wqt = vbf + XE;           // transposed bf16 weights (k-major)
  __bf16* wkt = wqt + WE;
  __bf16* wvt = wkt + WE;
  __bf16* wot = wvt + WE;
  __bf16* Qh  = wot + WE;           // (B,H,S,dk)
  __bf16* Kt  = Qh + XE;            // (B,H,dk,S)
  __bf16* Vh  = Kt + XE;            // (B,H,S,dk)
  __bf16* Oh  = qbf;                // reuse after projections: (B,S,D)

  float* outp = (float*)d_out;
  float* attn = outp + XE;

  const dim3 cblk(256);
  cast_bf16_kernel<<<dim3((unsigned)((XE + 255) / 256)), cblk, 0, stream>>>(q, qbf, (int)XE);
  cast_bf16_kernel<<<dim3((unsigned)((XE + 255) / 256)), cblk, 0, stream>>>(k, kbf, (int)XE);
  cast_bf16_kernel<<<dim3((unsigned)((XE + 255) / 256)), cblk, 0, stream>>>(v, vbf, (int)XE);

  const dim3 tgrid(D_MODEL / 32, D_MODEL / 32);
  transpose_cast_kernel<<<tgrid, cblk, 0, stream>>>(Wq, wqt, D_MODEL, D_MODEL);
  transpose_cast_kernel<<<tgrid, cblk, 0, stream>>>(Wk, wkt, D_MODEL, D_MODEL);
  transpose_cast_kernel<<<tgrid, cblk, 0, stream>>>(Wv, wvt, D_MODEL, D_MODEL);
  transpose_cast_kernel<<<tgrid, cblk, 0, stream>>>(Wo, wot, D_MODEL, D_MODEL);

  const dim3 gblk(128);
  // projections: M=4096, N=1024, K=1024
  gemm_wmma_nt<0, false><<<dim3(D_MODEL / BN, MTOT / BM, 1), gblk, 0, stream>>>(
      qbf, wqt, bq, nullptr, Qh, MTOT, D_MODEL, D_MODEL, 0.f);
  gemm_wmma_nt<1, false><<<dim3(D_MODEL / BN, MTOT / BM, 1), gblk, 0, stream>>>(
      kbf, wkt, bk, nullptr, Kt, MTOT, D_MODEL, D_MODEL, 0.f);
  gemm_wmma_nt<0, false><<<dim3(D_MODEL / BN, MTOT / BM, 1), gblk, 0, stream>>>(
      vbf, wvt, bv, nullptr, Vh, MTOT, D_MODEL, D_MODEL, 0.f);

  // scores = scale * Q K^T (+mask): A=Qh (S,dk), Bk=Kt (dk,S); M=N=2048, K=64
  gemm_wmma_nt<4, false><<<dim3(SEQ / BN, SEQ / BM, BATCH * NHEADS), gblk, 0, stream>>>(
      Qh, Kt, nullptr, mask, attn, SEQ, SEQ, DK, 0.125f);

  softmax_rows<<<dim3(BATCH * NHEADS * SEQ), dim3(256), 0, stream>>>(attn);

  // out_head = attn @ V: A=attn fp32, Bk=Vh (S,dk); M=2048, N=64, K=2048
  gemm_wmma_nt<5, true><<<dim3(1, SEQ / BM, BATCH * NHEADS), gblk, 0, stream>>>(
      attn, Vh, nullptr, nullptr, Oh, SEQ, DK, SEQ, 0.f);

  // final projection: out = Oh @ Wo^T + bo -> fp32
  gemm_wmma_nt<2, false><<<dim3(D_MODEL / BN, MTOT / BM, 1), gblk, 0, stream>>>(
      Oh, wot, bo, nullptr, outp, MTOT, D_MODEL, D_MODEL, 0.f);
}
